// GNNEncoder1_4389456576916
// MI455X (gfx1250) — matrix operand order repaired and checked by reading, not compiled, not run on hardware
//
#include <hip/hip_runtime.h>
#include <hip/hip_bf16.h>

// ---------------------------------------------------------------------------
// Types for CDNA5 WMMA
// ---------------------------------------------------------------------------
typedef __attribute__((ext_vector_type(16))) __bf16 v16bf;
typedef __attribute__((ext_vector_type(8)))  float  v8f;
typedef __attribute__((ext_vector_type(4)))  int    v4i;

union FragBF {
    uint4 u[2];   // 32 bytes = 16 bf16
    v16bf v;
};

__device__ __forceinline__ __bf16 f2bf(float f) {
    union { unsigned short s; __bf16 b; } u;
    unsigned x = __float_as_uint(f);
    u.s = (unsigned short)((x + 0x7fffu + ((x >> 16) & 1u)) >> 16); // RNE
    return u.b;
}

// monotone float <-> ordered-uint key for atomicMax on floats
__device__ __forceinline__ unsigned fkey(float f) {
    int i = __float_as_int(f);
    return (i >= 0) ? (((unsigned)i) ^ 0x80000000u) : ~((unsigned)i);
}
__device__ __forceinline__ float fdec(unsigned k) {
    int i = (k & 0x80000000u) ? (int)(k ^ 0x80000000u) : (int)~k;
    return __int_as_float(i);
}

__device__ __forceinline__ int imin(int a, int b) { return a < b ? a : b; }

#define HID   128
#define CTXW  2048
#define INW   2176   // HID + CTXW
#define HC    512    // HEADS*HID
#define KPRBF 736    // 720 padded to /32
#define NGRAPH 64

// GEMM block tile
#define BM 128
#define BN 128
#define BK 32
#define LSH 40       // padded LDS row stride in bf16 (breaks bank conflicts)

// ----- async-to-LDS support (guarded; falls back to sync staging) ----------
#if defined(__has_builtin)
#if __has_builtin(__builtin_amdgcn_global_load_async_to_lds_b128) && \
    __has_builtin(__builtin_amdgcn_s_wait_asynccnt)
#define HAVE_ASYNC 1
#endif
#endif
#ifndef HAVE_ASYNC
#define HAVE_ASYNC 0
#endif

#if HAVE_ASYNC
typedef __attribute__((address_space(1))) v4i v4i_g;
typedef __attribute__((address_space(3))) v4i v4i_l;
__device__ __forceinline__ void async16(void* lds, const void* g) {
    // flat LDS address carries the LDS offset in addr[31:0]; the integer cast
    // truncates to the 32-bit AS3 offset, which is exactly what the HW wants.
    __builtin_amdgcn_global_load_async_to_lds_b128(
        (v4i_g*)(size_t)g,
        (v4i_l*)(size_t)lds,
        0, 0);
}
#endif

// ---------------------------------------------------------------------------
// Block-tiled bf16 WMMA GEMM:  C[M,Nc] = A[M,K] * BT[Nc,K]^T  (+ epilogues)
// 256 threads = 8 waves; block computes BM x BN; wave w computes rows
// [w*16, w*16+16) of the block tile across all BN columns. Per K-step the
// wave loads its A fragment and ALL 8 B fragments into distinct registers
// before issuing the 8 WMMAs, so the LDS reads overlap and the matrix ops
// run back-to-back. A/B K-panels staged in LDS, double-buffered with async
// global->LDS copies.
// Epilogues (enabled by non-null pointers):
//   bias[n]; ctx[batch[m]*Nc+n]; BN(eval)+ReLU; f32 store; bf16 store;
//   pool: atomicAdd(pool[batch[m]*Nc+n]).
// ---------------------------------------------------------------------------
__global__ __launch_bounds__(256)
void k_gemm_bf16(const __bf16* __restrict__ A, int lda,
                 const __bf16* __restrict__ BT, int ldb,
                 int M, int Nc, int K,
                 const float* __restrict__ bias,
                 const float* __restrict__ ctx, const int* __restrict__ batch,
                 const float* __restrict__ bng, const float* __restrict__ bnb,
                 const float* __restrict__ bnm, const float* __restrict__ bnv,
                 float* __restrict__ Cf, int ldc,
                 __bf16* __restrict__ Cbf, int ldcb, int coloff,
                 float* __restrict__ pool)
{
    __shared__ __align__(16) __bf16 Ash[2][BM * LSH];
    __shared__ __align__(16) __bf16 Bsh[2][BN * LSH];

    const int tid  = threadIdx.x;
    const int lane = tid & 31;
    const int w    = tid >> 5;
    const int nt   = Nc >> 7;                    // 128-wide N tiles
    const int bm   = blockIdx.x / nt;
    const int bn   = blockIdx.x % nt;
    const int tmb  = bm << 7;
    const int tnb  = bn << 7;

    const int r  = lane & 15;
    const int hi = lane >> 4;                    // lane half

    // staging indices: each thread moves 2 A chunks + 2 B chunks of 16B
    const int srow0 = tid >> 2;                  // 0..63
    const int srow1 = srow0 + 64;                // 64..127
    const int skoff = (tid & 3) << 3;            // {0,8,16,24} bf16
    const int ga0 = imin(tmb + srow0, M - 1);    // clamp ragged M
    const int ga1 = imin(tmb + srow1, M - 1);

    v8f acc[8];
#pragma unroll
    for (int j = 0; j < 8; ++j)
        acc[j] = (v8f){0.f,0.f,0.f,0.f,0.f,0.f,0.f,0.f};

#if HAVE_ASYNC
    // per-thread staging pointers, advanced by BK each K-step
    const __bf16* pa0 = A  + (size_t)ga0 * lda + skoff;
    const __bf16* pa1 = A  + (size_t)ga1 * lda + skoff;
    const __bf16* pb0 = BT + (size_t)(tnb + srow0) * ldb + skoff;
    const __bf16* pb1 = BT + (size_t)(tnb + srow1) * ldb + skoff;
    __bf16* const lA0[2] = { &Ash[0][srow0 * LSH + skoff], &Ash[1][srow0 * LSH + skoff] };
    __bf16* const lA1[2] = { &Ash[0][srow1 * LSH + skoff], &Ash[1][srow1 * LSH + skoff] };
    __bf16* const lB0[2] = { &Bsh[0][srow0 * LSH + skoff], &Bsh[1][srow0 * LSH + skoff] };
    __bf16* const lB1[2] = { &Bsh[0][srow1 * LSH + skoff], &Bsh[1][srow1 * LSH + skoff] };

    // prime buffer 0
    async16(lA0[0], pa0);  async16(lA1[0], pa1);
    async16(lB0[0], pb0);  async16(lB1[0], pb1);
    pa0 += BK; pa1 += BK; pb0 += BK; pb1 += BK;

    int buf = 0;
    for (int k0 = 0; k0 < K; k0 += BK) {
        __builtin_amdgcn_s_wait_asynccnt(0);
        __syncthreads();
        if (k0 + BK < K) {
            const int nb = buf ^ 1;
            async16(lA0[nb], pa0);  async16(lA1[nb], pa1);
            async16(lB0[nb], pb0);  async16(lB1[nb], pb1);
            pa0 += BK; pa1 += BK; pb0 += BK; pb1 += BK;
        }
        // load A fragment + all 8 B fragments into live registers first
        FragBF a, bb[8];
        const __bf16* arp   = &Ash[buf][((w << 4) + r) * LSH + (hi << 3)];
        const __bf16* bbase = &Bsh[buf][r * LSH + (hi << 3)];
        a.u[0] = *(const uint4*)(arp);
        a.u[1] = *(const uint4*)(arp + 16);
#pragma unroll
        for (int j = 0; j < 8; ++j) {
            const __bf16* br = bbase + (size_t)(j << 4) * LSH;
            bb[j].u[0] = *(const uint4*)(br);
            bb[j].u[1] = *(const uint4*)(br + 16);
        }
#pragma unroll
        for (int j = 0; j < 8; ++j)
            acc[j] = __builtin_amdgcn_wmma_f32_16x16x32_bf16(
                         false, a.v, false, bb[j].v, (short)0, acc[j], false, false);
        buf ^= 1;
    }
#else
    // ---- synchronous LDS staging fallback ----
    for (int k0 = 0; k0 < K; k0 += BK) {
        uint4 a0 = *(const uint4*)(A  + (size_t)ga0 * lda + k0 + skoff);
        uint4 a1 = *(const uint4*)(A  + (size_t)ga1 * lda + k0 + skoff);
        uint4 b0 = *(const uint4*)(BT + (size_t)(tnb + srow0) * ldb + k0 + skoff);
        uint4 b1 = *(const uint4*)(BT + (size_t)(tnb + srow1) * ldb + k0 + skoff);
        __syncthreads();
        *(uint4*)&Ash[0][srow0 * LSH + skoff] = a0;
        *(uint4*)&Ash[0][srow1 * LSH + skoff] = a1;
        *(uint4*)&Bsh[0][srow0 * LSH + skoff] = b0;
        *(uint4*)&Bsh[0][srow1 * LSH + skoff] = b1;
        __syncthreads();
        FragBF a;
        const __bf16* ar = &Ash[0][((w << 4) + r) * LSH + (hi << 3)];
        a.u[0] = *(const uint4*)(ar);
        a.u[1] = *(const uint4*)(ar + 16);
#pragma unroll
        for (int j = 0; j < 8; ++j) {
            FragBF b;
            const __bf16* br = &Bsh[0][((j << 4) + r) * LSH + (hi << 3)];
            b.u[0] = *(const uint4*)(br);
            b.u[1] = *(const uint4*)(br + 16);
            acc[j] = __builtin_amdgcn_wmma_f32_16x16x32_bf16(
                         false, a.v, false, b.v, (short)0, acc[j], false, false);
        }
    }
#endif

    // Epilogue.  M = tmb + w*16 + rr + 8*hi,  N = tnb + j*16 + r
    const int mbase = tmb + (w << 4) + (hi << 3);
#pragma unroll
    for (int j = 0; j < 8; ++j) {
        const int n = tnb + (j << 4) + r;
        const float bi = bias ? bias[n] : 0.f;
        float sc = 1.f, sh = 0.f;
        if (bng) { sc = bng[n] * rsqrtf(bnv[n] + 1e-5f); sh = bnb[n] - bnm[n] * sc; }
#pragma unroll
        for (int rr = 0; rr < 8; ++rr) {
            const int m = mbase + rr;
            if (m >= M) continue;
            float v = acc[j][rr] + bi;
            if (ctx)  v += ctx[(size_t)batch[m] * Nc + n];
            if (bng)  { v = fmaf(v, sc, sh); v = fmaxf(v, 0.f); }
            if (pool) atomicAdd(&pool[(size_t)batch[m] * Nc + n], v);
            if (Cf)   Cf[(size_t)m * ldc + n] = v;
            if (Cbf)  Cbf[(size_t)m * ldcb + coloff + n] = f2bf(v);
        }
    }
}

// ---------------------------------------------------------------------------
// Weight convert + transpose:  WT[n*Kp + k] = bf16(W[k*Nc + n]), zero-pad k>=K
// ---------------------------------------------------------------------------
__global__ void k_wt(const float* __restrict__ W, __bf16* __restrict__ WT,
                     int K, int Nc, int Kp)
{
    size_t i = (size_t)blockIdx.x * blockDim.x + threadIdx.x;
    if (i >= (size_t)Nc * Kp) return;
    int n = (int)(i / Kp), k = (int)(i % Kp);
    float v = (k < K) ? W[(size_t)k * Nc + n] : 0.f;
    WT[i] = f2bf(v);
}

__global__ void k_fill_u32(unsigned* __restrict__ p, size_t n, unsigned v)
{
    size_t i = (size_t)blockIdx.x * blockDim.x + threadIdx.x;
    if (i < n) p[i] = v;
}

// x[n,c] = emb_W[atom[n],c] + emb_b[c]; also bf16 into xc cols [0,128)
__global__ void k_node_init(const int* __restrict__ atom,
                            const float* __restrict__ embW,
                            const float* __restrict__ embb,
                            float* __restrict__ x, __bf16* __restrict__ xc)
{
    int n = blockIdx.x, c = threadIdx.x;
    float v = embW[(size_t)atom[n] * HID + c] + embb[c];
    x[(size_t)n * HID + c] = v;
    xc[(size_t)n * INW + c] = f2bf(v);
}

// rbf[n,k] = exp(-0.5*(pos[n,dim]-mu)^2) (720 real cols, pad to 736)
__global__ void k_rbf(const float* __restrict__ pos, __bf16* __restrict__ rbf, int N)
{
    size_t idx = (size_t)blockIdx.x * blockDim.x + threadIdx.x;
    if (idx >= (size_t)N * KPRBF) return;
    int n = (int)(idx / KPRBF), k = (int)(idx % KPRBF);
    float v = 0.f;
    if (k < 720) {
        int dim = k / 240, bin = k % 240;
        float mu = -30.f + 60.f * (float)bin / 239.f;
        float d  = pos[(size_t)n * 3 + dim] - mu;
        v = __expf(-0.5f * d * d);
    }
    rbf[idx] = f2bf(v);
}

// attention scores: one wave per (node, head); 128-wide dot, wave reduce
__global__ void k_att(const float* __restrict__ xs,
                      const float* __restrict__ attS,
                      const float* __restrict__ attD,
                      float* __restrict__ a_src, float* __restrict__ a_dst, int N)
{
    int w = (int)((blockIdx.x * (size_t)blockDim.x + threadIdx.x) >> 5);
    int lane = threadIdx.x & 31;
    if (w >= N * 4) return;
    int n = w >> 2, h = w & 3;
    const float* xp = xs + (size_t)n * HC + h * HID;
    float s = 0.f, d = 0.f;
#pragma unroll
    for (int c = 0; c < HID; c += 32) {
        float v = xp[c + lane];
        s = fmaf(v, attS[h * HID + c + lane], s);
        d = fmaf(v, attD[h * HID + c + lane], d);
    }
#pragma unroll
    for (int o = 16; o > 0; o >>= 1) {
        s += __shfl_xor(s, o, 32);
        d += __shfl_xor(d, o, 32);
    }
    if (lane == 0) { a_src[w] = s; a_dst[w] = d; }
}

// leaky-relu(a_src[s]+a_dst[d]) and segment-max via ordered-key atomicMax
__global__ void k_alpha1(const int* __restrict__ ei,
                         const float* __restrict__ a_src,
                         const float* __restrict__ a_dst,
                         float* __restrict__ alpha, unsigned* __restrict__ amax,
                         int E, int N)
{
    size_t idx = (size_t)blockIdx.x * blockDim.x + threadIdx.x;
    if (idx >= (size_t)(E + N) * 4) return;
    int e = (int)(idx >> 2), h = (int)(idx & 3);
    int s = (e < E) ? ei[e]     : (e - E);
    int d = (e < E) ? ei[E + e] : (e - E);
    float v = a_src[s * 4 + h] + a_dst[d * 4 + h];
    v = (v > 0.f) ? v : 0.2f * v;
    alpha[idx] = v;
    atomicMax(&amax[d * 4 + h], fkey(v));
}

// alpha = exp(alpha - amax[dst]); denom[dst] += alpha
__global__ void k_alpha2(const int* __restrict__ ei,
                         float* __restrict__ alpha,
                         const unsigned* __restrict__ amax,
                         float* __restrict__ denom, int E, int N)
{
    size_t idx = (size_t)blockIdx.x * blockDim.x + threadIdx.x;
    if (idx >= (size_t)(E + N) * 4) return;
    int e = (int)(idx >> 2), h = (int)(idx & 3);
    int d = (e < E) ? ei[E + e] : (e - E);
    float v = __expf(alpha[idx] - fdec(amax[d * 4 + h]));
    alpha[idx] = v;
    atomicAdd(&denom[d * 4 + h], v);
}

// message scatter: agg[dst] += xs[src] * softmax weight ; one block per edge
__global__ void k_msg(const int* __restrict__ ei,
                      const float* __restrict__ xs,
                      const float* __restrict__ alpha,
                      const float* __restrict__ denom,
                      float* __restrict__ agg, int E, int N)
{
    int e = blockIdx.x;
    int s = (e < E) ? ei[e]     : (e - E);
    int d = (e < E) ? ei[E + e] : (e - E);
    for (int c = threadIdx.x; c < HC; c += blockDim.x) {
        int h = c >> 7;
        float w = alpha[(size_t)e * 4 + h] / fmaxf(denom[d * 4 + h], 1e-16f);
        atomicAdd(&agg[(size_t)d * HC + c], xs[(size_t)s * HC + c] * w);
    }
}

// head-mean + bias + BN + exact GELU + residual; refresh bf16 in xc cols [0,128)
__global__ void k_post(const float* __restrict__ agg, const float* __restrict__ gbias,
                       const float* __restrict__ g, const float* __restrict__ b,
                       const float* __restrict__ m, const float* __restrict__ v,
                       float* __restrict__ x, __bf16* __restrict__ xc)
{
    int n = blockIdx.x, c = threadIdx.x;
    const float* ap = agg + (size_t)n * HC;
    float a = 0.25f * (ap[c] + ap[128 + c] + ap[256 + c] + ap[384 + c]) + gbias[c];
    float sc = g[c] * rsqrtf(v[c] + 1e-5f);
    a = fmaf(a - m[c], sc, b[c]);
    float ge = 0.5f * a * (1.f + erff(a * 0.70710678118f));
    float xn = x[(size_t)n * HID + c] + ge;
    x[(size_t)n * HID + c] = xn;
    xc[(size_t)n * INW + c] = f2bf(xn);
}

// ---------------------------------------------------------------------------
// Host orchestration
// ---------------------------------------------------------------------------
static inline size_t cdiv(size_t a, size_t b) { return (a + b - 1) / b; }

extern "C" void kernel_launch(void* const* d_in, const int* in_sizes, int n_in,
                              void* d_out, int out_size, void* d_ws, size_t ws_size,
                              hipStream_t stream)
{
    const int*   atom   = (const int*)  d_in[0];
    const float* pos    = (const float*)d_in[1];
    const int*   ei     = (const int*)  d_in[2];
    const int*   batch  = (const int*)  d_in[4];
    const float* ctxvec = (const float*)d_in[5];
    const float* embW   = (const float*)d_in[6];
    const float* embb   = (const float*)d_in[7];
    const float* r1W    = (const float*)d_in[8];
    const float* r1b    = (const float*)d_in[9];
    const float* rbg    = (const float*)d_in[10];
    const float* rbb    = (const float*)d_in[11];
    const float* rbm    = (const float*)d_in[12];
    const float* rbv    = (const float*)d_in[13];
    const float* r2W    = (const float*)d_in[14];
    const float* r2b    = (const float*)d_in[15];
    const float* gatW   = (const float*)d_in[16];
    const float* attS   = (const float*)d_in[17];
    const float* attD   = (const float*)d_in[18];
    const float* gbias  = (const float*)d_in[19];
    const float* bng    = (const float*)d_in[20];
    const float* bnb    = (const float*)d_in[21];
    const float* bnm    = (const float*)d_in[22];
    const float* bnv    = (const float*)d_in[23];
    const float* outW   = (const float*)d_in[24];
    const float* outb   = (const float*)d_in[25];

    const int N = in_sizes[0];
    const int E = in_sizes[2] / 2;

    // workspace carve-out (256B aligned)
    char* cur = (char*)d_ws;
    auto alloc = [&](size_t bytes) -> void* {
        void* p = (void*)cur;
        cur += (bytes + 255) & ~(size_t)255;
        return p;
    };
    __bf16* xc    = (__bf16*)alloc((size_t)N * INW   * 2);
    __bf16* rbf   = (__bf16*)alloc((size_t)N * KPRBF * 2);
    __bf16* h1    = (__bf16*)alloc((size_t)N * HID   * 2);
    float*  x     = (float*) alloc((size_t)N * HID   * 4);
    float*  xs    = (float*) alloc((size_t)N * HC    * 4);
    float*  agg   = (float*) alloc((size_t)N * HC    * 4);
    float*  a_src = (float*) alloc((size_t)N * 4 * 4);
    float*  a_dst = (float*) alloc((size_t)N * 4 * 4);
    float*  alpha = (float*) alloc((size_t)(E + N) * 4 * 4);
    unsigned* amax= (unsigned*)alloc((size_t)N * 4 * 4);
    float*  denom = (float*) alloc((size_t)N * 4 * 4);
    __bf16* r1WT  = (__bf16*)alloc((size_t)HID  * KPRBF * 2);
    __bf16* r2WT  = (__bf16*)alloc((size_t)CTXW * HID   * 2);
    __bf16* gWT   = (__bf16*)alloc((size_t)3 * HC * INW * 2);
    __bf16* oWT   = (__bf16*)alloc((size_t)1024 * HID   * 2);

    auto fill0 = [&](void* p, size_t words) {
        k_fill_u32<<<(unsigned)cdiv(words, 256), 256, 0, stream>>>((unsigned*)p, words, 0u);
    };
    auto gemm = [&](const __bf16* A, int lda, const __bf16* BT, int ldb,
                    int M, int Nc, int K, const float* bias,
                    const float* ctx, const int* bat,
                    const float* g, const float* b, const float* m, const float* v,
                    float* Cf, int ldc, __bf16* Cbf, int ldcb, int coloff, float* pool) {
        unsigned blocks = (unsigned)(cdiv(M, BM) * (size_t)(Nc / BN));
        k_gemm_bf16<<<blocks, 256, 0, stream>>>(
            A, lda, BT, ldb, M, Nc, K, bias, ctx, bat, g, b, m, v,
            Cf, ldc, Cbf, ldcb, coloff, pool);
    };

    // zero output accumulator (harness poisons it)
    fill0(d_out, (size_t)NGRAPH * 1024);

    // weight conversion (f32 -> bf16, transposed, K padded to /32)
    k_wt<<<(unsigned)cdiv((size_t)HID * KPRBF, 256), 256, 0, stream>>>(r1W, r1WT, 720, HID, KPRBF);
    k_wt<<<(unsigned)cdiv((size_t)CTXW * HID, 256), 256, 0, stream>>>(r2W, r2WT, HID, CTXW, HID);
    for (int l = 0; l < 3; ++l)
        k_wt<<<(unsigned)cdiv((size_t)HC * INW, 256), 256, 0, stream>>>(
            gatW + (size_t)l * INW * HC, gWT + (size_t)l * HC * INW, INW, HC, INW);
    k_wt<<<(unsigned)cdiv((size_t)1024 * HID, 256), 256, 0, stream>>>(outW, oWT, HID, 1024, HID);

    // node features + RBF features
    k_node_init<<<N, HID, 0, stream>>>(atom, embW, embb, x, xc);
    k_rbf<<<(unsigned)cdiv((size_t)N * KPRBF, 256), 256, 0, stream>>>(pos, rbf, N);

    // RBF MLP:  h1 = relu(BN(rbf @ W1 + b1))   [bf16 out]
    gemm(rbf, KPRBF, r1WT, KPRBF, N, HID, KPRBF, r1b,
         nullptr, nullptr, rbg, rbb, rbm, rbv,
         nullptr, 0, h1, HID, 0, nullptr);
    // context = h1 @ W2 + b2 + ctxvec[batch]   -> bf16 into xc cols [128,2176)
    gemm(h1, HID, r2WT, HID, N, CTXW, HID, r2b,
         ctxvec, batch, nullptr, nullptr, nullptr, nullptr,
         nullptr, 0, xc, INW, HID, nullptr);

    // GAT layers
    for (int l = 0; l < 3; ++l) {
        gemm(xc, INW, gWT + (size_t)l * HC * INW, INW, N, HC, INW,
             nullptr, nullptr, nullptr, nullptr, nullptr, nullptr, nullptr,
             xs, HC, nullptr, 0, 0, nullptr);

        k_att<<<(unsigned)cdiv((size_t)N * 4 * 32, 256), 256, 0, stream>>>(
            xs, attS + (size_t)l * 4 * HID, attD + (size_t)l * 4 * HID, a_src, a_dst, N);

        fill0(amax, (size_t)N * 4);
        fill0(denom, (size_t)N * 4);
        fill0(agg, (size_t)N * HC);

        size_t na = (size_t)(E + N) * 4;
        k_alpha1<<<(unsigned)cdiv(na, 256), 256, 0, stream>>>(ei, a_src, a_dst, alpha, amax, E, N);
        k_alpha2<<<(unsigned)cdiv(na, 256), 256, 0, stream>>>(ei, alpha, amax, denom, E, N);
        k_msg<<<E + N, 256, 0, stream>>>(ei, xs, alpha, denom, agg, E, N);

        k_post<<<N, HID, 0, stream>>>(agg, gbias + (size_t)l * HID,
                                      bng + (size_t)l * HID, bnb + (size_t)l * HID,
                                      bnm + (size_t)l * HID, bnv + (size_t)l * HID,
                                      x, xc);
    }

    // out = x @ out_W + out_b, pooled by graph via atomics into d_out
    gemm(xc, INW, oWT, HID, N, 1024, HID, outb,
         nullptr, batch, nullptr, nullptr, nullptr, nullptr,
         nullptr, 0, nullptr, 0, 0, (float*)d_out);

    (void)n_in; (void)out_size; (void)ws_size;
}
